// SoftEquivariantLayer_37898791420143
// MI455X (gfx1250) — matrix operand order repaired and checked by reading, not compiled, not run on hardware
//
#include <hip/hip_runtime.h>
#include <math.h>
#include <stdint.h>

#define NB   8
#define BD   16
#define TD   128
#define HID  64
#define ROWS_PER_BLOCK 256
#define THREADS 256
#define EPSF 1e-8f

// padded LDS row stride for B tiles: 144 bf16 = 288 B (32B-aligned rows,
// per-lane bank offset of 8 -> conflict-free-at-b128-rate ds reads)
#define BSTRIDE 144
#define SBUF_BYTES (2 * TD * BSTRIDE * 2)   // hi + lo matrices = 73728 B

typedef __attribute__((ext_vector_type(16))) __bf16 v16bf;
typedef __attribute__((ext_vector_type(8)))  float  v8f;

// ---------------------------------------------------------------------------
// Prep: untangle mixW [n,n,d,k] -> Mlin[r=i*16+d][c=j*16+k]; split fp32 into
// bf16 hi + bf16 lo. ws: hi at [0,16384), lo at [16384,32768). 64 KB total.
// ---------------------------------------------------------------------------
__global__ __launch_bounds__(256) void prep_mixw_kernel(
    const float* __restrict__ mixW, __bf16* __restrict__ ws) {
  int idx = blockIdx.x * blockDim.x + threadIdx.x;
  if (idx >= TD * TD) return;
  int r = idx >> 7, c = idx & 127;
  int i = r >> 4, d = r & 15, j = c >> 4, k = c & 15;
  float x = mixW[((i * NB + j) * BD + d) * BD + k];
  __bf16 h = (__bf16)x;
  __bf16 l = (__bf16)(x - (float)h);
  ws[idx]           = h;
  ws[TD * TD + idx] = l;
}

__device__ __forceinline__ float gelu_exact(float x) {
  return 0.5f * x * (1.0f + erff(x * 0.70710678118654752f));
}

// ---------------------------------------------------------------------------
__global__ __launch_bounds__(THREADS) void soft_equiv_kernel(
    const float* __restrict__ z,
    const float* __restrict__ W1, const float* __restrict__ b1,
    const float* __restrict__ W2, const float* __restrict__ b2,
    const float* __restrict__ W3, const float* __restrict__ gate_bias,
    const __bf16* __restrict__ Mhi,   // ws: hi then lo (lo = Mhi + TD*TD)
    float* __restrict__ out) {
  __shared__ float sW1[HID * NB];
  __shared__ float sb1[HID];
  __shared__ float sW2[HID * HID];
  __shared__ float sb2[HID];
  __shared__ float sW3[NB * HID];
  __shared__ float sScale[ROWS_PER_BLOCK][NB];
  // time-shared buffer: phase 1 = per-thread h1 scratch (256*68*4 = 69632 B),
  // phase 2 = padded bf16 B tiles (73728 B)
  __shared__ __align__(32) unsigned char sBuf[SBUF_BYTES];

  const int tid = threadIdx.x;
  const int blockRow = blockIdx.x * ROWS_PER_BLOCK;

  // ---- stage MLP weights into LDS (broadcast reads later) ----
  for (int i = tid; i < HID * NB;  i += THREADS) sW1[i] = W1[i];
  for (int i = tid; i < HID;       i += THREADS) { sb1[i] = b1[i]; sb2[i] = b2[i]; }
  for (int i = tid; i < HID * HID; i += THREADS) sW2[i] = W2[i];
  for (int i = tid; i < NB * HID;  i += THREADS) sW3[i] = W3[i];
  __syncthreads();

  // ---- phase 1: one row per thread: bundle norms -> MLP -> softplus scales ----
  {
    const float* zr = z + (size_t)(blockRow + tid) * TD;
    float norms[NB];
#pragma unroll
    for (int bdl = 0; bdl < NB; ++bdl) {
      float s = 0.f;
#pragma unroll
      for (int q = 0; q < 4; ++q) {
        float4 p = *(const float4*)(zr + bdl * BD + q * 4);
        s = fmaf(p.x, p.x, s); s = fmaf(p.y, p.y, s);
        s = fmaf(p.z, p.z, s); s = fmaf(p.w, p.w, s);
      }
      norms[bdl] = sqrtf(s) + EPSF;
    }
    float* h1 = (float*)sBuf + tid * 68;   // padded stride vs 64: bank spread
    for (int o = 0; o < HID; ++o) {
      float acc = sb1[o];
#pragma unroll
      for (int i = 0; i < NB; ++i) acc = fmaf(norms[i], sW1[o * NB + i], acc);
      h1[o] = gelu_exact(acc);
    }
    float sacc[NB];
#pragma unroll
    for (int j2 = 0; j2 < NB; ++j2) sacc[j2] = 0.f;
    for (int o = 0; o < HID; ++o) {
      float acc = sb2[o];
      for (int i = 0; i < HID; ++i) acc = fmaf(h1[i], sW2[o * HID + i], acc);
      float h2 = gelu_exact(acc);
#pragma unroll
      for (int j2 = 0; j2 < NB; ++j2) sacc[j2] = fmaf(h2, sW3[j2 * HID + o], sacc[j2]);
    }
#pragma unroll
    for (int j2 = 0; j2 < NB; ++j2) {
      float x = sacc[j2];                      // stable softplus
      sScale[tid][j2] = (x > 20.0f) ? x : log1pf(expf(x));
    }
  }
  __syncthreads();   // h1 scratch dead from here; sBuf becomes B staging

  // ---- async stage B (bf16 hi/lo, padded rows) from ws into LDS ----
  // 4096 x 16B chunks total, 16 per thread, CDNA5 async global->LDS path.
  {
    const unsigned ldsB = (unsigned)(uintptr_t)&sBuf[0];  // LDS byte offset
#pragma unroll
    for (int it = 0; it < 16; ++it) {
      const int chunk = tid + it * THREADS;   // 0..4095
      const int mat   = chunk >> 11;          // 0 = hi, 1 = lo
      const int rem   = chunk & 2047;
      const int row   = rem >> 4;             // 0..127
      const int c16   = rem & 15;             // 16B chunk within row
      unsigned dsaddr = ldsB + (unsigned)(mat * (TD * BSTRIDE * 2) +
                                          row * (BSTRIDE * 2) + c16 * 16);
      unsigned goff   = (unsigned)(mat * (TD * TD * 2) + row * (TD * 2) + c16 * 16);
      asm volatile("global_load_async_to_lds_b128 %0, %1, %2"
                   :: "v"(dsaddr), "v"(goff), "s"(Mhi) : "memory");
    }
    asm volatile("s_wait_asynccnt 0x0" ::: "memory");
  }
  __syncthreads();

  // ---- phase 2: per-wave WMMA over two 16-row tiles ----
  const int wave = tid >> 5;
  const int lane = tid & 31;
  const int m16  = lane & 15;
  const int half = lane >> 4;
  const __bf16* sB = (const __bf16*)sBuf;

  // A (z rows), bf16 hi/lo split in contiguous-k order (same order as B): lane l holds row (l&15);
  // elements 0..15 = K kb..kb+15, kb = kt*32 + (l>>4)*16.
  v16bf aHi[2][4], aLo[2][4];
#pragma unroll
  for (int rt = 0; rt < 2; ++rt) {
    const int rowInBlk = (wave * 2 + rt) * 16 + m16;
    const float* zr = z + (size_t)(blockRow + rowInBlk) * TD;
#pragma unroll
    for (int kt = 0; kt < 4; ++kt) {
      const int kb = kt * 32 + half * 16;
      float v[16];
      float4 p;
      p = *(const float4*)(zr + kb);      v[0]=p.x;  v[1]=p.y;  v[2]=p.z;  v[3]=p.w;
      p = *(const float4*)(zr + kb + 4);  v[4]=p.x;  v[5]=p.y;  v[6]=p.z;  v[7]=p.w;
      p = *(const float4*)(zr + kb + 8);  v[8]=p.x;  v[9]=p.y;  v[10]=p.z; v[11]=p.w;
      p = *(const float4*)(zr + kb + 12); v[12]=p.x; v[13]=p.y; v[14]=p.z; v[15]=p.w;
#pragma unroll
      for (int e = 0; e < 16; ++e) {
        __bf16 h = (__bf16)v[e];
        aHi[rt][kt][e] = h;
        aLo[rt][kt][e] = (__bf16)(v[e] - (float)h);
      }
    }
  }

  for (int nt = 0; nt < NB; ++nt) {
    v8f acc0 = {};
    v8f acc1 = {};
    // one-stage software pipeline on B: load kt+1 while WMMAing kt
    const unsigned bbase = (unsigned)((nt * 16 + m16) * BSTRIDE + half * 16);
    v16bf bHi = *(const v16bf*)(sB + bbase);
    v16bf bLo = *(const v16bf*)(sB + TD * BSTRIDE + bbase);
#pragma unroll
    for (int kt = 0; kt < 4; ++kt) {
      const v16bf curHi = bHi;
      const v16bf curLo = bLo;
      if (kt < 3) {
        const unsigned bo = bbase + (unsigned)((kt + 1) * 32);
        bHi = *(const v16bf*)(sB + bo);
        bLo = *(const v16bf*)(sB + TD * BSTRIDE + bo);
      }
      // error-compensated fp32-via-bf16: hi*hi + lo*hi + hi*lo, f32 accumulate
      acc0 = __builtin_amdgcn_wmma_f32_16x16x32_bf16(false, aHi[0][kt], false, curHi, (short)0, acc0, false, false);
      acc0 = __builtin_amdgcn_wmma_f32_16x16x32_bf16(false, aLo[0][kt], false, curHi, (short)0, acc0, false, false);
      acc0 = __builtin_amdgcn_wmma_f32_16x16x32_bf16(false, aHi[0][kt], false, curLo, (short)0, acc0, false, false);
      acc1 = __builtin_amdgcn_wmma_f32_16x16x32_bf16(false, aHi[1][kt], false, curHi, (short)0, acc1, false, false);
      acc1 = __builtin_amdgcn_wmma_f32_16x16x32_bf16(false, aLo[1][kt], false, curHi, (short)0, acc1, false, false);
      acc1 = __builtin_amdgcn_wmma_f32_16x16x32_bf16(false, aHi[1][kt], false, curLo, (short)0, acc1, false, false);
    }
    // Fused epilogue: columns nt*16.. belong to bundle nt:
    // out = z*(1+scale[nt]) + sigmoid(gb[nt])*mixed, C/D layout
    // (VGPR v -> row v + (l>>4)*8, col l&15).
    const float g = __builtin_amdgcn_rcpf(1.0f + __expf(-gate_bias[nt]));
#pragma unroll
    for (int rt = 0; rt < 2; ++rt) {
      v8f& acc = rt ? acc1 : acc0;
#pragma unroll
      for (int v = 0; v < 8; ++v) {
        const int rowInBlk = (wave * 2 + rt) * 16 + v + half * 8;
        const int col = nt * 16 + m16;
        const size_t gi = (size_t)(blockRow + rowInBlk) * TD + col;
        const float zv = z[gi];
        const float sc = sScale[rowInBlk][nt];
        out[gi] = fmaf(zv, sc, zv) + g * acc[v];
      }
    }
  }
}

// ---------------------------------------------------------------------------
extern "C" void kernel_launch(void* const* d_in, const int* in_sizes, int n_in,
                              void* d_out, int out_size, void* d_ws, size_t ws_size,
                              hipStream_t stream) {
  const float* z    = (const float*)d_in[0];
  const float* W1   = (const float*)d_in[1];
  const float* b1   = (const float*)d_in[2];
  const float* W2   = (const float*)d_in[3];
  const float* b2   = (const float*)d_in[4];
  const float* W3   = (const float*)d_in[5];
  const float* mixW = (const float*)d_in[6];
  const float* gb   = (const float*)d_in[7];
  float*   out = (float*)d_out;
  __bf16*  ws  = (__bf16*)d_ws;   // needs 2*128*128*2 = 64 KB

  (void)in_sizes; (void)n_in; (void)ws_size;

  // 1) split mixW into bf16 hi/lo in B-friendly linear layout
  prep_mixw_kernel<<<(TD * TD + 255) / 256, 256, 0, stream>>>(mixW, ws);

  // 2) fused norms/MLP/scales + async B staging + split-WMMA mix + epilogue
  const int rows   = out_size / TD;            // 524288
  const int blocks = rows / ROWS_PER_BLOCK;    // 2048
  soft_equiv_kernel<<<blocks, THREADS, 0, stream>>>(
      z, W1, b1, W2, b2, W3, gb, ws, out);
}
